// GRULayer_74302934221406
// MI455X (gfx1250) — matrix operand order repaired and checked by reading, not compiled, not run on hardware
//
#include <hip/hip_runtime.h>

// GRU: B=64, T=512, I=256, H=512 (gates r,z,n packed as 3H=1536)
#define B_  64
#define T_  512
#define I_  256
#define H_  512
#define G3_ 1536

typedef __attribute__((ext_vector_type(16))) _Float16 v16h;
typedef __attribute__((ext_vector_type(8)))  float    v8f;

union V16H { v16h v; unsigned u[8]; };

// ---- WMMA operand loaders (layouts per CDNA5 ISA 7.12.2) --------------------
// A-matrix 16x32 f16, row-major source with leading dim `ld` (elements).
// lane L (0..31): row = L&15; lanes 0-15 hold K {0..7, 16..23}, lanes 16-31
// hold K {8..15, 24..31}; each VGPR holds 2 consecutive K halves -> dword load.
__device__ __forceinline__ v16h load_a16x32(const _Float16* base, int ld, int lane){
  const int r = lane & 15, hs = lane >> 4;
  V16H t;
#pragma unroll
  for (int q = 0; q < 8; ++q){
    const int k = ((q >> 2) << 4) + (hs << 3) + ((q & 3) << 1);
    t.u[q] = *reinterpret_cast<const unsigned*>(base + r * ld + k);
  }
  return t.v;
}

// B-matrix 32x16 f16 where B[k][n] = W[n][k], W row-major [N,K] with ld.
// lane L: column n = L&15 (i.e. W row n); lanes 0-15 hold K {0..7, 16..23}, lanes
// 16-31 hold K {8..15, 24..31}, 2 consecutive K per VGPR -> dword load.
__device__ __forceinline__ v16h load_b32x16(const _Float16* base, int ld, int lane){
  const int n = lane & 15, hs = lane >> 4;
  V16H t;
#pragma unroll
  for (int q = 0; q < 8; ++q){
    const int k = ((q >> 2) << 4) + (hs << 3) + ((q & 3) << 1);
    t.u[q] = *reinterpret_cast<const unsigned*>(base + n * ld + k);
  }
  return t.v;
}

// ---- fp32 -> fp16 conversion -----------------------------------------------
__global__ void cvt_f32_f16(const float* __restrict__ s, _Float16* __restrict__ d, int n){
  int i = blockIdx.x * blockDim.x + threadIdx.x;
  if (i < n) d[i] = (_Float16)s[i];
}

// ---- Phase 1: x_proj[B*T, 3H] = x f16-GEMM W_ih^T + b_ih (stored f16) ------
// grid = (2048 M-tiles, 12), block = 256 (8 waves); wave w owns N-tile
// blockIdx.y*8+w. K = 256 -> 8 chunks of 32. No guards: EXEC all-1s at WMMA.
__global__ void xproj_gemm(const _Float16* __restrict__ Xh,
                           const _Float16* __restrict__ Wih,
                           const float*    __restrict__ bih,
                           _Float16*       __restrict__ XP){
  const int lane = threadIdx.x & 31;
  const int wave = threadIdx.x >> 5;
  const int mt = blockIdx.x;
  const int nt = blockIdx.y * 8 + wave;          // 0..95
  const _Float16* abase = Xh  + (size_t)(mt * 16) * I_;
  const _Float16* bbase = Wih + (size_t)(nt * 16) * I_;
  v8f c = {0.f,0.f,0.f,0.f,0.f,0.f,0.f,0.f};
#pragma unroll
  for (int kc = 0; kc < I_ / 32; ++kc){
    v16h a = load_a16x32(abase + kc * 32, I_, lane);
    v16h b = load_b32x16(bbase + kc * 32, I_, lane);
    c = __builtin_amdgcn_wmma_f32_16x16x32_f16(false, a, false, b, (short)0, c, false, false);
  }
  // C/D layout: element (m,n): vgpr r = m&7, lane = (n&15) + 16*(m>=8)
  const int n  = lane & 15;
  const int mb = (lane >> 4) * 8;
  const int col = nt * 16 + n;
  const float bias = bih[col];
#pragma unroll
  for (int r = 0; r < 8; ++r){
    const int row = mt * 16 + mb + r;
    XP[(size_t)row * G3_ + col] = (_Float16)(c[r] + bias);
  }
}

// ---- Phase 2: persistent per-batch-slice scan ------------------------------
// 4 blocks x 1024 threads (32 waves); block owns 16 batch rows. Recurrent
// state h lives entirely in LDS (f32 + f16 mirror); per step:
//   pass A: h_proj[16,1536] = h_f16 WMMA W_hh^T  (each wave: 3 N-tiles x 16 K-chunks)
//           (FUSED: also x_proj tiles from x_f16 WMMA W_ih^T)
//   pass B: elementwise GRU gates, update h in LDS.
// Only __syncthreads between passes — no grid sync (batches independent).
template<bool FUSED>
__global__ void __launch_bounds__(1024) gru_scan(
    const _Float16* __restrict__ XPh,   // precomputed x_proj f16 (!FUSED)
    const _Float16* __restrict__ Xh,    // x f16 [B*T, I]        (FUSED)
    const _Float16* __restrict__ Wih,   // [3H, I] f16           (FUSED)
    const _Float16* __restrict__ Whh,   // [3H, H] f16
    const float*    __restrict__ bih,
    const float*    __restrict__ bhh,
    float*          __restrict__ out)   // [B, H] f32
{
  __shared__ _Float16 hh[16 * H_];              // 16 KB  h (f16, WMMA A operand)
  __shared__ float    hf[16 * H_];              // 32 KB  h (f32, exact state)
  __shared__ float    hp[16 * G3_];             // 96 KB  h-projection
  __shared__ float    xp[FUSED ? 16 * G3_ : 4]; // 96 KB  x-projection (fused only)

  const int tid  = threadIdx.x;
  const int lane = tid & 31;
  const int wave = tid >> 5;                    // 0..31
  const int b0   = blockIdx.x * 16;             // batch-slice base

  for (int i = tid; i < 16 * H_; i += 1024){ hf[i] = 0.f; hh[i] = (_Float16)0.f; }
  __syncthreads();

  for (int t = 0; t < T_; ++t){
    // ---- pass A: projections (96 N-tiles over 32 waves -> 3 each) ----
#pragma unroll
    for (int tt = 0; tt < 3; ++tt){
      const int nt = wave * 3 + tt;
      const _Float16* bbase = Whh + (size_t)(nt * 16) * H_;
      v8f c = {0.f,0.f,0.f,0.f,0.f,0.f,0.f,0.f};
#pragma unroll 4
      for (int kc = 0; kc < H_ / 32; ++kc){
        v16h a = load_a16x32(hh + kc * 32, H_, lane);       // LDS -> ds loads
        v16h b = load_b32x16(bbase + kc * 32, H_, lane);
        c = __builtin_amdgcn_wmma_f32_16x16x32_f16(false, a, false, b, (short)0, c, false, false);
      }
      const int n  = lane & 15;
      const int mb = (lane >> 4) * 8;
#pragma unroll
      for (int r = 0; r < 8; ++r) hp[(mb + r) * G3_ + nt * 16 + n] = c[r];

      if (FUSED){
        const _Float16* abase = Xh  + (size_t)(b0 * T_ + t) * I_;  // row stride T_*I_
        const _Float16* bb2   = Wih + (size_t)(nt * 16) * I_;
        v8f cx = {0.f,0.f,0.f,0.f,0.f,0.f,0.f,0.f};
#pragma unroll
        for (int kc = 0; kc < I_ / 32; ++kc){
          v16h a = load_a16x32(abase + kc * 32, T_ * I_, lane);
          v16h b = load_b32x16(bb2 + kc * 32, I_, lane);
          cx = __builtin_amdgcn_wmma_f32_16x16x32_f16(false, a, false, b, (short)0, cx, false, false);
        }
#pragma unroll
        for (int r = 0; r < 8; ++r) xp[(mb + r) * G3_ + nt * 16 + n] = cx[r];
      }
    }
    __syncthreads();

    // ---- pass B: gates over 16*512 elements (8 per thread) ----
    for (int e = tid; e < 16 * H_; e += 1024){
      const int bb = e >> 9;            // local batch 0..15
      const int j  = e & (H_ - 1);      // hidden index
      float xr, xz, xn;
      if (FUSED){
        xr = xp[bb * G3_ +           j] + bih[j];
        xz = xp[bb * G3_ + H_ +      j] + bih[H_ + j];
        xn = xp[bb * G3_ + 2 * H_ +  j] + bih[2 * H_ + j];
      } else {
        const _Float16* xrow = XPh + (size_t)((b0 + bb) * T_ + t) * G3_;
        xr = (float)xrow[j];
        xz = (float)xrow[H_ + j];
        xn = (float)xrow[2 * H_ + j];
        if (t + 1 < T_ && (j & 31) == 0)   // one prefetch per 64B line
          __builtin_prefetch(XPh + (size_t)((b0 + bb) * T_ + t + 1) * G3_ + j, 0, 0);
      }
      const float hr = hp[bb * G3_ +          j] + bhh[j];
      const float hz = hp[bb * G3_ + H_ +     j] + bhh[H_ + j];
      const float hn = hp[bb * G3_ + 2 * H_ + j] + bhh[2 * H_ + j];
      const float r  = 1.f / (1.f + __expf(-(xr + hr)));
      const float z  = 1.f / (1.f + __expf(-(xz + hz)));
      const float nn = tanhf(xn + r * hn);
      const float hnew = (1.f - z) * nn + z * hf[e];
      hf[e] = hnew;
      hh[e] = (_Float16)hnew;
    }
    __syncthreads();
  }

  for (int e = tid; e < 16 * H_; e += 1024){
    const int bb = e >> 9, j = e & (H_ - 1);
    out[(size_t)(b0 + bb) * H_ + j] = hf[e];
  }
}

// ---- host-side launcher ----------------------------------------------------
extern "C" void kernel_launch(void* const* d_in, const int* in_sizes, int n_in,
                              void* d_out, int out_size, void* d_ws, size_t ws_size,
                              hipStream_t stream) {
  const float* x   = (const float*)d_in[0];   // [64, 512, 256]
  const float* Wih = (const float*)d_in[1];   // [1536, 256]
  const float* Whh = (const float*)d_in[2];   // [1536, 512]
  const float* bih = (const float*)d_in[3];   // [1536]
  const float* bhh = (const float*)d_in[4];   // [1536]
  float* out = (float*)d_out;                 // [64, 512]

  const size_t nX   = (size_t)B_ * T_ * I_;   // 8,388,608
  const size_t nWih = (size_t)G3_ * I_;       //   393,216
  const size_t nWhh = (size_t)G3_ * H_;       //   786,432
  const size_t nXP  = (size_t)B_ * T_ * G3_;  // 50,331,648

  char* ws = (char*)d_ws;
  size_t off = 0;
  auto carve = [&](size_t bytes) -> char* {
    char* p = ws + off;
    off = (off + bytes + 255) & ~(size_t)255;
    return p;
  };
  _Float16* xh   = (_Float16*)carve(nX   * 2);
  _Float16* wihh = (_Float16*)carve(nWih * 2);
  _Float16* whhh = (_Float16*)carve(nWhh * 2);
  const size_t need_min  = off;
  _Float16* xph  = (_Float16*)carve(nXP  * 2);
  const size_t need_full = off;

  const bool precomp = (ws_size >= need_full);
  (void)need_min;

  cvt_f32_f16<<<(int)((nX   + 255) / 256), 256, 0, stream>>>(x,   xh,   (int)nX);
  cvt_f32_f16<<<(int)((nWih + 255) / 256), 256, 0, stream>>>(Wih, wihh, (int)nWih);
  cvt_f32_f16<<<(int)((nWhh + 255) / 256), 256, 0, stream>>>(Whh, whhh, (int)nWhh);

  if (precomp){
    dim3 grid(B_ * T_ / 16, G3_ / 16 / 8);    // (2048, 12), 8 waves/block
    xproj_gemm<<<grid, 256, 0, stream>>>(xh, wihh, bih, xph);
    gru_scan<false><<<B_ / 16, 1024, 0, stream>>>(xph, nullptr, nullptr, whhh, bih, bhh, out);
  } else {
    gru_scan<true><<<B_ / 16, 1024, 0, stream>>>(nullptr, xh, wihh, whhh, bih, bhh, out);
  }
}
